// KGAtt_19610820674273
// MI455X (gfx1250) — matrix operand order repaired and checked by reading, not compiled, not run on hardware
//
#include <hip/hip_runtime.h>

typedef __attribute__((ext_vector_type(16))) _Float16 v16h;
typedef __attribute__((ext_vector_type(8)))  float    v8f;

#define E_EDGES 500000
#define N_ENTS  50000
#define IN_DIM  128
#define OUT_DIM 128
#define K_STEPS 12           // 384 / 32
#define N_TILES 8            // 128 / 16
#define LRELU_SLOPE 0.01f
#define LDS_HALFS (K_STEPS * N_TILES * 512)   // 96 fragments * 32 lanes * 16 halves

__global__ void kg_zero(float* __restrict__ w, int n) {
  int i = blockIdx.x * blockDim.x + threadIdx.x;
  if (i < n) w[i] = 0.0f;
}

// Build a 16-half A fragment from one 128-float feature row.
// p points at row + k2*32 + hi*8; group0 = p[0..7], group1 = p[16..23].
__device__ __forceinline__ v16h make_a(const float* __restrict__ p) {
  float4 g0a = ((const float4*)p)[0];
  float4 g0b = ((const float4*)p)[1];
  float4 g1a = ((const float4*)(p + 16))[0];
  float4 g1b = ((const float4*)(p + 16))[1];
  v16h a;
  a[0]  = (_Float16)g0a.x; a[1]  = (_Float16)g0a.y;
  a[2]  = (_Float16)g0a.z; a[3]  = (_Float16)g0a.w;
  a[4]  = (_Float16)g0b.x; a[5]  = (_Float16)g0b.y;
  a[6]  = (_Float16)g0b.z; a[7]  = (_Float16)g0b.w;
  a[8]  = (_Float16)g1a.x; a[9]  = (_Float16)g1a.y;
  a[10] = (_Float16)g1a.z; a[11] = (_Float16)g1a.w;
  a[12] = (_Float16)g1b.x; a[13] = (_Float16)g1b.y;
  a[14] = (_Float16)g1b.z; a[15] = (_Float16)g1b.w;
  return a;
}

// One K segment of 128 (4 WMMA K-steps) with a statically known feature source.
// B fragment is double-buffered (1-ahead) and reused across both M-tiles.
template <int SEG>
__device__ __forceinline__ void do_segment(const float* __restrict__ rowA0,
                                           const float* __restrict__ rowA1,
                                           const _Float16* __restrict__ ldsB,
                                           int lane, int hi,
                                           v8f acc0[N_TILES], v8f acc1[N_TILES]) {
#pragma unroll
  for (int k2 = 0; k2 < 4; ++k2) {
    const int kk = SEG * 4 + k2;
    const int lo = k2 * 32 + hi * 8;
    v16h a0 = make_a(rowA0 + lo);
    v16h a1 = make_a(rowA1 + lo);
    const _Float16* bbase = ldsB + (size_t)(kk * N_TILES) * 512 + lane * 16;

    v16h bcur = *(const v16h*)(bbase);
#pragma unroll
    for (int nt = 0; nt < N_TILES; ++nt) {
      v16h bnext;
      if (nt < N_TILES - 1) bnext = *(const v16h*)(bbase + (nt + 1) * 512);
      acc0[nt] = __builtin_amdgcn_wmma_f32_16x16x32_f16(
          false, a0, false, bcur, (short)0, acc0[nt], false, false);
      acc1[nt] = __builtin_amdgcn_wmma_f32_16x16x32_f16(
          false, a1, false, bcur, (short)0, acc1[nt], false, false);
      bcur = bnext;
    }
  }
}

// Epilogue for one 16-edge tile: bias, attention logit, exp(leaky), scatter.
__device__ __forceinline__ void tile_epilogue(v8f acc[N_TILES],
                                              const int* __restrict__ trip,
                                              int ebase, int mlow, int hi,
                                              const float* __restrict__ w2v,
                                              const float* __restrict__ bav,
                                              float b2,
                                              float* __restrict__ numer,
                                              float* __restrict__ denom) {
#pragma unroll
  for (int nt = 0; nt < N_TILES; ++nt)
#pragma unroll
    for (int r = 0; r < 8; ++r) acc[nt][r] += bav[nt];

  float part[8];
#pragma unroll
  for (int r = 0; r < 8; ++r) {
    float p = 0.0f;
#pragma unroll
    for (int nt = 0; nt < N_TILES; ++nt) p += acc[nt][r] * w2v[nt];
    part[r] = p;
  }
  // Butterfly reduce within the 16-lane half that owns these rows.
#pragma unroll
  for (int mask = 1; mask <= 8; mask <<= 1)
#pragma unroll
    for (int r = 0; r < 8; ++r) part[r] += __shfl_xor(part[r], mask, 32);

  float eb[8];
#pragma unroll
  for (int r = 0; r < 8; ++r) {
    float s = part[r] + b2;
    float lr = (s > 0.0f) ? s : LRELU_SLOPE * s;
    eb[r] = __expf(lr);
  }

#pragma unroll
  for (int r = 0; r < 8; ++r) {
    const int srcm = trip[ebase + hi * 8 + r];
    float* nrow = numer + (size_t)srcm * OUT_DIM;
#pragma unroll
    for (int nt = 0; nt < N_TILES; ++nt)
      atomicAdd(nrow + nt * 16 + mlow, eb[r] * acc[nt][r]);
    if (mlow == 0) atomicAdd(denom + srcm, eb[r]);
  }
}

__global__ void __launch_bounds__(256)
kg_att_main(const int*   __restrict__ trip,   // [3, E]
            const float* __restrict__ ent,    // [N, 128]
            const float* __restrict__ rel,    // [R, 128]
            const float* __restrict__ Wa,     // [384, 128] row-major
            const float* __restrict__ ba,     // [128]
            const float* __restrict__ Wa2,    // [128]
            const float* __restrict__ ba2,    // [1]
            float* __restrict__ numer,        // [N, 128]  (atomic accum)
            float* __restrict__ denom,        // [N]       (atomic accum)
            int nPairs) {
  extern __shared__ __align__(32) _Float16 ldsB[];

  const int tid = threadIdx.x;

  // ---- Stage W_a (f32 -> f16) into LDS in WMMA B-fragment layout.
  //   k = kk*32 + (j<8 ? 8*(l>>4) + j : 16 + 8*(l>>4) + j - 8),  n = nt*16 + (l&15)
  for (int idx = tid; idx < LDS_HALFS; idx += 256) {
    int f   = idx >> 9;        // fragment 0..95
    int rem = idx & 511;
    int l   = rem >> 4;        // lane 0..31
    int j   = rem & 15;        // half 0..15
    int kk  = f >> 3;
    int nt  = f & 7;
    int k   = kk * 32 + ((j < 8) ? (8 * (l >> 4) + j) : (16 + 8 * (l >> 4) + j - 8));
    int n   = nt * 16 + (l & 15);
    ldsB[idx] = (_Float16)Wa[k * OUT_DIM + n];
  }
  __syncthreads();

  const int wave = blockIdx.x * (blockDim.x >> 5) + (tid >> 5);
  if (wave >= nPairs) return;              // uniform per wave -> EXEC all-ones

  const int lane = tid & 31;
  const int mlow = lane & 15;
  const int hi   = lane >> 4;
  const int eb0  = wave * 32;              // tile0 edges [eb0, eb0+16)
  const int eb1  = eb0 + 16;               // tile1 edges [eb1, eb1+16)

  // Per-lane A-row feature pointers for both tiles (statically segmented).
  const int e0 = eb0 + mlow, e1 = eb1 + mlow;
  const float* rs0 = ent + (size_t)trip[e0] * IN_DIM;
  const float* rr0 = rel + (size_t)trip[E_EDGES + e0] * IN_DIM;
  const float* rd0 = ent + (size_t)trip[2 * E_EDGES + e0] * IN_DIM;
  const float* rs1 = ent + (size_t)trip[e1] * IN_DIM;
  const float* rr1 = rel + (size_t)trip[E_EDGES + e1] * IN_DIM;
  const float* rd1 = ent + (size_t)trip[2 * E_EDGES + e1] * IN_DIM;

  v8f acc0[N_TILES], acc1[N_TILES];
  {
    v8f z = {};
#pragma unroll
    for (int nt = 0; nt < N_TILES; ++nt) { acc0[nt] = z; acc1[nt] = z; }
  }

  do_segment<0>(rs0, rs1, ldsB, lane, hi, acc0, acc1);   // K 0..127   (ent[src])
  do_segment<1>(rr0, rr1, ldsB, lane, hi, acc0, acc1);   // K 128..255 (rel)
  do_segment<2>(rd0, rd1, ldsB, lane, hi, acc0, acc1);   // K 256..383 (ent[dst])

  float w2v[N_TILES], bav[N_TILES];
#pragma unroll
  for (int nt = 0; nt < N_TILES; ++nt) {
    int n = nt * 16 + mlow;
    w2v[nt] = Wa2[n];
    bav[nt] = ba[n];
  }
  const float b2 = ba2[0];

  tile_epilogue(acc0, trip, eb0, mlow, hi, w2v, bav, b2, numer, denom);
  tile_epilogue(acc1, trip, eb1, mlow, hi, w2v, bav, b2, numer, denom);
}

__global__ void kg_finalize(const float* __restrict__ numer,
                            const float* __restrict__ denom,
                            float* __restrict__ out, int n) {
  int i = blockIdx.x * blockDim.x + threadIdx.x;
  if (i >= n) return;
  float v = numer[i] / denom[i >> 7];                 // row = i / 128
  out[i] = (v > 0.0f) ? v : (__expf(v) - 1.0f);       // elu
}

extern "C" void kernel_launch(void* const* d_in, const int* in_sizes, int n_in,
                              void* d_out, int out_size, void* d_ws, size_t ws_size,
                              hipStream_t stream) {
  const int*   trip = (const int*)d_in[0];
  const float* ent  = (const float*)d_in[1];
  const float* rel  = (const float*)d_in[2];
  // d_in[3] = adj (unused by forward)
  const float* Wa   = (const float*)d_in[4];
  const float* ba   = (const float*)d_in[5];
  const float* Wa2  = (const float*)d_in[6];
  const float* ba2  = (const float*)d_in[7];

  float* numer = (float*)d_ws;                         // N * 128 floats
  float* denom = numer + (size_t)N_ENTS * OUT_DIM;     // N floats

  const int zn = N_ENTS * OUT_DIM + N_ENTS;
  kg_zero<<<(zn + 255) / 256, 256, 0, stream>>>(numer, zn);

  const int nPairs = E_EDGES / 32;                     // 15,625 exactly
  const int blocks = (nPairs + 7) / 8;                 // 8 waves per block
  const size_t ldsBytes = (size_t)LDS_HALFS * sizeof(_Float16);   // 96 KB
  kg_att_main<<<blocks, 256, ldsBytes, stream>>>(trip, ent, rel, Wa, ba, Wa2, ba2,
                                                 numer, denom, nPairs);

  const int on = N_ENTS * OUT_DIM;
  kg_finalize<<<(on + 255) / 256, 256, 0, stream>>>(numer, denom, (float*)d_out, on);
}
